// MultiHeadAttention_7095285973829
// MI455X (gfx1250) — matrix operand order repaired and checked by reading, not compiled, not run on hardware
//
#include <hip/hip_runtime.h>
#include <hip/hip_bf16.h>

typedef __attribute__((ext_vector_type(16))) _Float16 v16h;
typedef __attribute__((ext_vector_type(8)))  float    v8f;

union AF { v16h h; unsigned u[8]; };
union BF { v16h h; unsigned u[8]; };

// ---------------------------------------------------------------------------
// Problem constants (reference: B=2, N=2048, E=1024, H=16, HD=64)
// ---------------------------------------------------------------------------
#define NB   2
#define NSEQ 2048
#define EMB  1024
#define NH   16
#define HD   64
#define MROWS (NB * NSEQ)      // 4096 flattened rows

// ---------------------------------------------------------------------------
// CDNA5 async global->LDS copy (ASYNCcnt-tracked DMA, ISA ch.8 / ch.10)
// LDS dst must be 16B aligned for b128. Low 32 bits of a generic pointer to
// __shared__ are the wave-relative LDS byte offset (aperture mapping).
// ---------------------------------------------------------------------------
__device__ __forceinline__ void async_load_b128(void* lds_dst, const void* gaddr) {
    asm volatile("global_load_async_to_lds_b128 %0, %1, off"
                 :: "v"((unsigned)(uintptr_t)lds_dst),
                    "v"((unsigned long long)(uintptr_t)gaddr)
                 : "memory");
}
__device__ __forceinline__ void wait_async0() {
    asm volatile("s_wait_asynccnt 0x0" ::: "memory");
}

// ---------------------------------------------------------------------------
// fp32 -> f16 conversion (grid-stride)
// ---------------------------------------------------------------------------
__global__ __launch_bounds__(256) void cvt_f32_f16(const float* __restrict__ src,
                                                   _Float16* __restrict__ dst, int n) {
    int i = blockIdx.x * blockDim.x + threadIdx.x;
    int stride = gridDim.x * blockDim.x;
    for (; i < n; i += stride) dst[i] = (_Float16)src[i];
}

// ---------------------------------------------------------------------------
// GEMM 1: qkv = x @ W_qkv + b_qkv, scattered into Q[b,h,n,d], K[b,h,n,d],
// Vt[b,h,d,n] (V transposed so the P@V B-fragment gather is contiguous).
// Block = 256 thr (8 waves), tile 128x64, K-step 32, f16 WMMA / f32 accum.
// A tile staged via async-to-LDS DMA; B tile staged transposed (coalesced
// u32 global reads, u16 LDS writes).
// ---------------------------------------------------------------------------
#define A_LD 40   // padded LDS row (halves); 80B rows keep b128 dst aligned,
                  // and 20*m mod 64 dword banks are distinct for m=0..15
#define B_LD 36

__global__ __launch_bounds__(256)
void qkv_gemm(const _Float16* __restrict__ xh,    // [4096][1024]
              const _Float16* __restrict__ wh,    // [1024][3072]
              const float* __restrict__ bias,     // [3072]
              _Float16* __restrict__ Qh, _Float16* __restrict__ Kh,
              _Float16* __restrict__ Vt) {
    constexpr int K = EMB, NC = 3 * EMB;
    __shared__ _Float16 As[128 * A_LD];
    __shared__ _Float16 Bs[64 * B_LD];

    const int tid  = threadIdx.x;
    const int lane = tid & 31;
    const int w    = tid >> 5;
    const int cbase = blockIdx.x * 64;
    const int rbase = blockIdx.y * 128;

    v8f acc[4] = {};
    const int m     = lane & 15;
    const int koffA = (lane < 16) ? 0 : 16;
    const int koffB = (lane < 16) ? 0 : 16;

    for (int kt = 0; kt < K; kt += 32) {
        // A tile: 128x32 halves, async DMA in 16B chunks (4 per row)
        for (int i = tid; i < 128 * 4; i += 256) {
            int r = i >> 2, c = i & 3;
            async_load_b128(&As[r * A_LD + 8 * c],
                            xh + (size_t)(rbase + r) * K + kt + 8 * c);
        }
        // B tile transposed into Bs[n][k]; coalesced u32 global reads
        for (int i = tid; i < 32 * 32; i += 256) {
            int np = i & 31, k = i >> 5;
            union { unsigned u; _Float16 f[2]; } t;
            t.u = *(const unsigned*)(wh + (size_t)(kt + k) * NC + cbase + 2 * np);
            Bs[(2 * np) * B_LD + k]     = t.f[0];
            Bs[(2 * np + 1) * B_LD + k] = t.f[1];
        }
        wait_async0();
        __syncthreads();

        AF a;
        const _Float16* arow = &As[(w * 16 + m) * A_LD];
#pragma unroll
        for (int v = 0; v < 8; ++v) {
            int k = 2 * v + koffA;
            a.u[v] = *(const unsigned*)(arow + k);
        }
#pragma unroll
        for (int nf = 0; nf < 4; ++nf) {
            BF bf;
            const _Float16* brow = &Bs[(nf * 16 + m) * B_LD];
#pragma unroll
            for (int v = 0; v < 8; ++v)
                bf.u[v] = *(const unsigned*)(brow + 2 * v + koffB);
            acc[nf] = __builtin_amdgcn_wmma_f32_16x16x32_f16(
                false, a.h, false, bf.h, (short)0, acc[nf], false, false);
        }
        __syncthreads();
    }

    // Epilogue: bias add, fold softmax scale into Q, scatter per-head
    const float qscale = 0.125f;  // HD^-0.5
    const int hi = (lane < 16) ? 0 : 8;
#pragma unroll
    for (int nf = 0; nf < 4; ++nf) {
        int col  = cbase + nf * 16 + m;
        float bv = bias[col];
        int head = col / (3 * HD);
        int j    = col % (3 * HD);
        int d    = j & (HD - 1);
        int which = j >> 6;  // 0=Q 1=K 2=V
#pragma unroll
        for (int r = 0; r < 8; ++r) {
            int grow = rbase + w * 16 + r + hi;
            int bi   = grow >> 11;
            int nrow = grow & (NSEQ - 1);
            float val = acc[nf][r] + bv;
            if (which == 0)
                Qh[((size_t)(bi * NH + head) * NSEQ + nrow) * HD + d] = (_Float16)(val * qscale);
            else if (which == 1)
                Kh[((size_t)(bi * NH + head) * NSEQ + nrow) * HD + d] = (_Float16)val;
            else
                Vt[((size_t)(bi * NH + head) * HD + d) * NSEQ + nrow] = (_Float16)val;
        }
    }
}

// ---------------------------------------------------------------------------
// Flash attention: each wave owns 16 query rows; streams 32 keys/iter.
// S = Q K^T via WMMA, online softmax, P bounced through per-wave LDS tile
// (C-layout -> A-layout), O += P V via WMMA against transposed V.
// ---------------------------------------------------------------------------
#define P_LD 34

__global__ __launch_bounds__(256)
void attention(const _Float16* __restrict__ Qh,  // [B][H][N][HD] (pre-scaled)
               const _Float16* __restrict__ Kh,  // [B][H][N][HD]
               const _Float16* __restrict__ Vt,  // [B][H][HD][N]
               _Float16* __restrict__ Oh) {      // [B][N][E]
    __shared__ _Float16 Pl[8 * 16 * P_LD];
    const int tid  = threadIdx.x;
    const int lane = tid & 31;
    const int w    = tid >> 5;
    const int h    = blockIdx.y;
    const int b    = blockIdx.z;
    const int qbase = blockIdx.x * 128 + w * 16;

    const size_t bh = (size_t)b * NH + h;
    const _Float16* Qp = Qh + bh * NSEQ * HD;
    const _Float16* Kp = Kh + bh * NSEQ * HD;
    const _Float16* Vp = Vt + bh * HD * NSEQ;

    const int m     = lane & 15;
    const int koffA = (lane < 16) ? 0 : 16;
    const int koffB = (lane < 16) ? 0 : 16;

    // Resident Q fragments (d-chunks 0..31, 32..63)
    AF qf[2];
#pragma unroll
    for (int c = 0; c < 2; ++c)
#pragma unroll
        for (int v = 0; v < 8; ++v) {
            int k = c * 32 + 2 * v + koffA;
            qf[c].u[v] = *(const unsigned*)(Qp + (size_t)(qbase + m) * HD + k);
        }

    v8f o[4] = {};
    float mrow[8], lrow[8];
#pragma unroll
    for (int r = 0; r < 8; ++r) { mrow[r] = -1e30f; lrow[r] = 0.f; }

    _Float16* Pw = &Pl[w * 16 * P_LD];

    for (int kb = 0; kb < NSEQ; kb += 32) {
        // S tiles: 16 rows x 32 keys (two 16x16 C fragments), contract over d=64
        v8f s[2] = {};
#pragma unroll
        for (int t = 0; t < 2; ++t) {
#pragma unroll
            for (int c = 0; c < 2; ++c) {
                BF kf;
                const _Float16* krow = Kp + (size_t)(kb + t * 16 + m) * HD + c * 32;
#pragma unroll
                for (int v = 0; v < 8; ++v)
                    kf.u[v] = *(const unsigned*)(krow + 2 * v + koffB);
                s[t] = __builtin_amdgcn_wmma_f32_16x16x32_f16(
                    false, qf[c].h, false, kf.h, (short)0, s[t], false, false);
            }
        }
        // Online softmax; row stats replicated across the 16 column lanes
#pragma unroll
        for (int r = 0; r < 8; ++r) {
            float mx = fmaxf(s[0][r], s[1][r]);
#pragma unroll
            for (int off = 8; off; off >>= 1)
                mx = fmaxf(mx, __shfl_xor(mx, off, 32));
            float mnew = fmaxf(mrow[r], mx);
            float corr = __expf(mrow[r] - mnew);
            float p0 = __expf(s[0][r] - mnew);
            float p1 = __expf(s[1][r] - mnew);
            float rs = p0 + p1;
#pragma unroll
            for (int off = 8; off; off >>= 1)
                rs += __shfl_xor(rs, off, 32);
            lrow[r] = lrow[r] * corr + rs;
            mrow[r] = mnew;
#pragma unroll
            for (int nf = 0; nf < 4; ++nf) o[nf][r] *= corr;
            int prow = r + ((lane < 16) ? 0 : 8);
            Pw[prow * P_LD + m]      = (_Float16)p0;
            Pw[prow * P_LD + 16 + m] = (_Float16)p1;
        }
        // Same-wave LDS write->read: per-wave LDS ops are in-order; fence DScnt
        asm volatile("s_wait_dscnt 0" ::: "memory");
        // Re-gather P in A-matrix layout
        AF pf;
#pragma unroll
        for (int v = 0; v < 8; ++v) {
            int k = 2 * v + koffA;
            pf.u[v] = *(const unsigned*)(&Pw[m * P_LD + k]);
        }
        // O += P @ V  (V transposed: key pairs contiguous -> u32 gathers)
#pragma unroll
        for (int nf = 0; nf < 4; ++nf) {
            BF vf;
            const _Float16* vrow = Vp + (size_t)(nf * 16 + m) * NSEQ + kb;
#pragma unroll
            for (int v = 0; v < 8; ++v)
                vf.u[v] = *(const unsigned*)(vrow + 2 * v + koffB);
            o[nf] = __builtin_amdgcn_wmma_f32_16x16x32_f16(
                false, pf.h, false, vf.h, (short)0, o[nf], false, false);
        }
    }

    // Normalize and store merged-head layout [B][N][E] as f16 for out-proj
#pragma unroll
    for (int r = 0; r < 8; ++r) {
        float inv = 1.0f / lrow[r];
        int row = qbase + r + ((lane < 16) ? 0 : 8);
#pragma unroll
        for (int nf = 0; nf < 4; ++nf) {
            int col = h * HD + nf * 16 + m;
            Oh[((size_t)b * NSEQ + row) * EMB + col] = (_Float16)(o[nf][r] * inv);
        }
    }
}

// ---------------------------------------------------------------------------
// GEMM 2: out = attn @ W_out + b_out, fp32 output
// ---------------------------------------------------------------------------
__global__ __launch_bounds__(256)
void out_gemm(const _Float16* __restrict__ ah,   // [4096][1024]
              const _Float16* __restrict__ wh,   // [1024][1024]
              const float* __restrict__ bias,    // [1024]
              float* __restrict__ out) {         // [4096][1024]
    constexpr int K = EMB, NC = EMB;
    __shared__ _Float16 As[128 * A_LD];
    __shared__ _Float16 Bs[64 * B_LD];

    const int tid  = threadIdx.x;
    const int lane = tid & 31;
    const int w    = tid >> 5;
    const int cbase = blockIdx.x * 64;
    const int rbase = blockIdx.y * 128;

    v8f acc[4] = {};
    const int m     = lane & 15;
    const int koffA = (lane < 16) ? 0 : 16;
    const int koffB = (lane < 16) ? 0 : 16;

    for (int kt = 0; kt < K; kt += 32) {
        for (int i = tid; i < 128 * 4; i += 256) {
            int r = i >> 2, c = i & 3;
            async_load_b128(&As[r * A_LD + 8 * c],
                            ah + (size_t)(rbase + r) * K + kt + 8 * c);
        }
        for (int i = tid; i < 32 * 32; i += 256) {
            int np = i & 31, k = i >> 5;
            union { unsigned u; _Float16 f[2]; } t;
            t.u = *(const unsigned*)(wh + (size_t)(kt + k) * NC + cbase + 2 * np);
            Bs[(2 * np) * B_LD + k]     = t.f[0];
            Bs[(2 * np + 1) * B_LD + k] = t.f[1];
        }
        wait_async0();
        __syncthreads();

        AF a;
        const _Float16* arow = &As[(w * 16 + m) * A_LD];
#pragma unroll
        for (int v = 0; v < 8; ++v) {
            int k = 2 * v + koffA;
            a.u[v] = *(const unsigned*)(arow + k);
        }
#pragma unroll
        for (int nf = 0; nf < 4; ++nf) {
            BF bf;
            const _Float16* brow = &Bs[(nf * 16 + m) * B_LD];
#pragma unroll
            for (int v = 0; v < 8; ++v)
                bf.u[v] = *(const unsigned*)(brow + 2 * v + koffB);
            acc[nf] = __builtin_amdgcn_wmma_f32_16x16x32_f16(
                false, a.h, false, bf.h, (short)0, acc[nf], false, false);
        }
        __syncthreads();
    }

    const int hi = (lane < 16) ? 0 : 8;
#pragma unroll
    for (int nf = 0; nf < 4; ++nf) {
        int col = cbase + nf * 16 + m;
        float bv = bias[col];
#pragma unroll
        for (int r = 0; r < 8; ++r) {
            int grow = rbase + w * 16 + r + hi;
            out[(size_t)grow * NC + col] = acc[nf][r] + bv;
        }
    }
}

// ---------------------------------------------------------------------------
// Launch
// ---------------------------------------------------------------------------
extern "C" void kernel_launch(void* const* d_in, const int* in_sizes, int n_in,
                              void* d_out, int out_size, void* d_ws, size_t ws_size,
                              hipStream_t stream) {
    const float* x    = (const float*)d_in[0];
    const float* Wqkv = (const float*)d_in[1];
    const float* bqkv = (const float*)d_in[2];
    const float* Wout = (const float*)d_in[3];
    const float* bout = (const float*)d_in[4];
    float* out = (float*)d_out;

    const size_t nX    = (size_t)MROWS * EMB;      // 4,194,304
    const size_t nWqkv = (size_t)EMB * 3 * EMB;    // 3,145,728
    const size_t nWout = (size_t)EMB * EMB;        // 1,048,576
    const size_t nQKV  = (size_t)NB * NH * NSEQ * HD;  // 4,194,304 each

    _Float16* ws    = (_Float16*)d_ws;
    _Float16* xh    = ws;
    _Float16* wqkvh = xh + nX;
    _Float16* wouth = wqkvh + nWqkv;
    _Float16* Qh    = wouth + nWout;
    _Float16* Kh    = Qh + nQKV;
    _Float16* Vt    = Kh + nQKV;
    _Float16* Oh    = Vt + nQKV;   // [B][N][E] f16, feeds out_gemm

    cvt_f32_f16<<<2048, 256, 0, stream>>>(x, xh, (int)nX);
    cvt_f32_f16<<<2048, 256, 0, stream>>>(Wqkv, wqkvh, (int)nWqkv);
    cvt_f32_f16<<<1024, 256, 0, stream>>>(Wout, wouth, (int)nWout);

    qkv_gemm<<<dim3(3 * EMB / 64, MROWS / 128), 256, 0, stream>>>(
        xh, wqkvh, bqkv, Qh, Kh, Vt);

    attention<<<dim3(NSEQ / 128, NH, NB), 256, 0, stream>>>(Qh, Kh, Vt, Oh);

    out_gemm<<<dim3(EMB / 64, MROWS / 128), 256, 0, stream>>>(Oh, wouth, bout, out);
}
